// IMTTB_14705968022080
// MI455X (gfx1250) — compile-verified, matrix operand order checked by reading, not run on hardware
//
#include <hip/hip_runtime.h>
#include <hip/hip_bf16.h>
#include <math.h>

// ---------------- types ----------------
typedef __bf16 bf16;
typedef __attribute__((ext_vector_type(16))) __bf16 v16bf;
typedef __attribute__((ext_vector_type(8)))  __bf16 v8bf;
typedef __attribute__((ext_vector_type(8)))  float  v8f;
typedef __attribute__((ext_vector_type(2)))  float  v2f;

#define HW   65536
#define IMG  256
#define NCH  64
#define CC   128
#define CHS  136          // padded channel stride (272B = multiple of 16B)
#define NPIX 324          // 18x18 halo pixels
#define NPAD 336          // rounded to 21 N-tiles of 16

// ---------------- helpers ----------------
// A-fragment (16x32 bf16) from f32 weights in global, row-major K-contiguous.
// Per ISA layout: lane holds M=lane&15, K = {8*half + 0..7, 16 + 8*half + 0..7} (+32*kc handled by caller).
__device__ __forceinline__ v16bf load_afrag(const float* p) {
  union { v16bf v; bf16 e[16]; } u;
#pragma unroll
  for (int i = 0; i < 8; ++i) {
    u.e[i]     = (bf16)p[i];
    u.e[i + 8] = (bf16)p[i + 16];
  }
  return u.v;
}

// B-fragment (32x16 bf16) from LDS stored [pixel][CHS] channel-contiguous.
// Lane holds N=lane&15; channels {8*half+0..7, 16+8*half+0..7}; caller passes row base + 8*half (+32*kc).
__device__ __forceinline__ v16bf load_bfrag(const bf16* p) {
  union { v16bf v; v8bf h[2]; } u;
  u.h[0] = *(const v8bf*)(p);
  u.h[1] = *(const v8bf*)(p + 16);
  return u.v;
}

// ---------------- kernel 1: zero ws accumulators ----------------
__global__ void init_kernel(float* p, int n) {
  int i = blockIdx.x * blockDim.x + threadIdx.x;
  if (i < n) p[i] = 0.0f;
}

// ---------------- kernel 2: Gram matrix + xx/yy partials ----------------
// grid (64 K-slabs, 8 batches), block 256 (8 waves).
// Wave w: M-tile m = w&3, N-tiles {2*(w>>2), 2*(w>>2)+1}; K loop over 1024 px in steps of 4
// using V_WMMA_F32_16X16X4_F32 (native f32 matrix op for the f32 matching reduction).
__global__ __launch_bounds__(256) void gram_kernel(
    const float* __restrict__ x, const float* __restrict__ ym,
    float* __restrict__ xy, float* __restrict__ xx, float* __restrict__ yy) {
  const int slab = blockIdx.x;
  const int b    = blockIdx.y;
  const int tid  = threadIdx.x;
  const int w    = tid >> 5, lane = tid & 31;
  const int m    = w & 3;
  const int n0   = (w >> 2) * 2;
  const int base = slab * 1024;

  const float* xb = x  + (size_t)b * NCH * HW;
  const float* yb = ym + (size_t)b * NCH * HW;

  const int ml    = lane & 15;
  const int half2 = (lane >> 4) * 2;
  const float* ax  = xb + (size_t)(16 * m  + ml) * HW + base + half2;
  const float* by0 = yb + (size_t)(16 * n0 + ml) * HW + base + half2;
  const float* by1 = by0 + (size_t)16 * HW;

  v8f acc0 = {}; v8f acc1 = {};
  for (int k = 0; k < 1024; k += 4) {
    v2f a  = *(const v2f*)(ax  + k);
    v2f bA = *(const v2f*)(by0 + k);
    v2f bB = *(const v2f*)(by1 + k);
    acc0 = __builtin_amdgcn_wmma_f32_16x16x4_f32(false, a, false, bA, (short)0, acc0, false, false);
    acc1 = __builtin_amdgcn_wmma_f32_16x16x4_f32(false, a, false, bB, (short)0, acc1, false, false);
  }
  const int rh = (lane >> 4) * 8;
#pragma unroll
  for (int r = 0; r < 8; ++r) {
    int Mch = 16 * m + r + rh;
    atomicAdd(&xy[((size_t)b * NCH + Mch) * NCH + 16 * n0 + ml],      acc0[r]);
    atomicAdd(&xy[((size_t)b * NCH + Mch) * NCH + 16 * n0 + 16 + ml], acc1[r]);
  }

  // xx / yy partial sums over this K-slab
  {
    int ch  = tid & 63;
    int sub = tid >> 6; // 0..3
    const float* px_ = xb + (size_t)ch * HW + base + sub * 256;
    const float* py_ = yb + (size_t)ch * HW + base + sub * 256;
    float sx = 0.f, sy = 0.f;
    for (int j = 0; j < 256; ++j) {
      float vx = px_[j], vy = py_[j];
      sx += vx * vx; sy += vy * vy;
    }
    atomicAdd(&xx[b * NCH + ch], sx);
    atomicAdd(&yy[b * NCH + ch], sy);
  }
}

// ---------------- kernel 3: per-channel nearest-neighbor argmin ----------------
__global__ void argmin_kernel(const float* __restrict__ xy, const float* __restrict__ xx,
                              const float* __restrict__ yy, int* __restrict__ nn) {
  int id = blockIdx.x * blockDim.x + threadIdx.x;
  if (id >= 8 * NCH) return;
  int b = id >> 6;
  float xc = xx[id];
  const float* row = xy + (size_t)id * NCH;
  float best = INFINITY; int bi = 0;
  for (int d = 0; d < NCH; ++d) {
    float v = xc + yy[b * NCH + d] - 2.0f * row[d];
    v = fmaxf(v, 0.0f);                 // matches reference clamp (ties -> first index)
    if (v < best) { best = v; bi = d; }
  }
  nn[id] = bi;
}

// ---------------- kernel 4: fused concat -> conv1x1 -> dwconv3x3+GELU -> conv1x1 -> (*concat) -> conv1x1 ----------------
// grid (16,16,8) tiles of 16x16 pixels; block 256 = 8 waves; everything stays in LDS.
__global__ __launch_bounds__(256) void fused_kernel(
    const float* __restrict__ x, const float* __restrict__ ym,
    const float* __restrict__ w1, const float* __restrict__ b1,
    const float* __restrict__ wd, const float* __restrict__ bd,
    const float* __restrict__ w2, const float* __restrict__ b2,
    const float* __restrict__ w12, const float* __restrict__ b12,
    const int* __restrict__ nn, float* __restrict__ out) {
  extern __shared__ char smem[];
  int*   sNN  = (int*)smem;                    // 64
  float* sB1  = (float*)(smem + 256);          // 128
  float* sBd  = sB1 + 128;                     // 128
  float* sB2  = sBd + 128;                     // 128
  float* sB12 = sB2 + 128;                     // 64
  float* sWd  = sB12 + 64;                     // 128*9
  bf16*  sC   = (bf16*)(smem + 6656);          // [NPAD][CHS] concat (halo), bf16
  bf16*  sH1  = sC + NPAD * CHS;               // [NPAD][CHS] h1, later reused for h4*concat
  bf16*  sH3  = sH1 + NPAD * CHS;              // [256][CHS]  gelu output

  const int tid = threadIdx.x;
  const int b   = blockIdx.z;
  const int tx0 = blockIdx.x * 16, ty0 = blockIdx.y * 16;

  // phase 0: stage small params
  if (tid < 64) { sNN[tid] = nn[b * NCH + tid]; sB12[tid] = b12[tid]; }
  if (tid < 128) { sB1[tid] = b1[tid]; sBd[tid] = bd[tid]; sB2[tid] = b2[tid]; }
  for (int i = tid; i < 128 * 9; i += 256) sWd[i] = wd[i];
  __syncthreads();

  // phase 1: gather concat = [x, Ym[nn]] into LDS (18x18 halo, coords clamped; taps outside
  // the image are masked to zero later in the depthwise stage, matching reference padding)
  {
    const float* xb = x  + (size_t)b * NCH * HW;
    const float* yb = ym + (size_t)b * NCH * HW;
    for (int c = 0; c < NCH; ++c) {
      const float* xc = xb + (size_t)c * HW;
      const float* yc = yb + (size_t)sNN[c] * HW;
      for (int p = tid; p < NPAD; p += 256) {
        bf16 vx = (bf16)0.0f, vy = (bf16)0.0f;
        if (p < NPIX) {
          int row = p / 18, col = p - row * 18;
          int gy = ty0 - 1 + row, gx = tx0 - 1 + col;
          gy = min(max(gy, 0), IMG - 1);
          gx = min(max(gx, 0), IMG - 1);
          vx = (bf16)xc[gy * IMG + gx];
          vy = (bf16)yc[gy * IMG + gx];
        }
        sC[p * CHS + c]       = vx;
        sC[p * CHS + NCH + c] = vy;
      }
    }
  }
  __syncthreads();

  const int w = tid >> 5, lane = tid & 31;
  const int ml = lane & 15, half = lane >> 4;

  // phase 2: GEMM1  h1 = W1(128x128) * concat(128 x 336px), accumulate f32, +b1, store bf16
  {
    v16bf a1[4];
    const float* Wr = w1 + (size_t)(16 * w + ml) * CC;
#pragma unroll
    for (int kc = 0; kc < 4; ++kc) a1[kc] = load_afrag(Wr + kc * 32 + 8 * half);

    for (int nt = 0; nt < 21; ++nt) {
      v8f acc = {};
#pragma unroll
      for (int kc = 0; kc < 4; ++kc) {
        v16bf bfrag = load_bfrag(sC + (nt * 16 + ml) * CHS + kc * 32 + 8 * half);
        acc = __builtin_amdgcn_wmma_f32_16x16x32_bf16(false, a1[kc], false, bfrag,
                                                      (short)0, acc, false, false);
      }
      int px = nt * 16 + ml;
#pragma unroll
      for (int r = 0; r < 8; ++r) {
        int ch = 16 * w + r + 8 * half;
        sH1[px * CHS + ch] = (bf16)(acc[r] + sB1[ch]);
      }
    }
  }
  __syncthreads();

  // phase 3: depthwise 3x3 (zero pad at image border) + exact GELU -> sH3
  {
    int pc = tid;                         // one center pixel per thread
    int cy = pc >> 4, cx = pc & 15;
    bool vmask[9]; int taddr[9];
#pragma unroll
    for (int dy = 0; dy < 3; ++dy)
#pragma unroll
      for (int dx = 0; dx < 3; ++dx) {
        int ry = cy + dy, rx = cx + dx;   // halo coords
        int gy = ty0 - 1 + ry, gx = tx0 - 1 + rx;
        vmask[dy * 3 + dx] = (gy >= 0 && gy < IMG && gx >= 0 && gx < IMG);
        taddr[dy * 3 + dx] = (ry * 18 + rx) * CHS;
      }
    for (int ch = 0; ch < CC; ++ch) {
      float acc = sBd[ch];
#pragma unroll
      for (int t = 0; t < 9; ++t)
        acc += vmask[t] ? sWd[ch * 9 + t] * (float)sH1[taddr[t] + ch] : 0.0f;
      float g = 0.5f * acc * (1.0f + erff(acc * 0.70710678118654752f));
      sH3[pc * CHS + ch] = (bf16)g;
    }
  }
  __syncthreads();

  // phase 4: GEMM2  h4 = W2(128x128) * h3, +b2, then elementwise * concat -> sH1 (reused)
  {
    v16bf a2[4];
    const float* Wr = w2 + (size_t)(16 * w + ml) * CC;
#pragma unroll
    for (int kc = 0; kc < 4; ++kc) a2[kc] = load_afrag(Wr + kc * 32 + 8 * half);

    for (int nt = 0; nt < 16; ++nt) {
      v8f acc = {};
#pragma unroll
      for (int kc = 0; kc < 4; ++kc) {
        v16bf bfrag = load_bfrag(sH3 + (nt * 16 + ml) * CHS + kc * 32 + 8 * half);
        acc = __builtin_amdgcn_wmma_f32_16x16x32_bf16(false, a2[kc], false, bfrag,
                                                      (short)0, acc, false, false);
      }
      int pc = nt * 16 + ml;
      int cy = pc >> 4, cx = pc & 15;
      int phalo = ((cy + 1) * 18 + (cx + 1)) * CHS;
#pragma unroll
      for (int r = 0; r < 8; ++r) {
        int ch = 16 * w + r + 8 * half;
        float h4v = acc[r] + sB2[ch];
        float cv  = (float)sC[phalo + ch];
        sH1[pc * CHS + ch] = (bf16)(h4v * cv);
      }
    }
  }
  __syncthreads();

  // phase 5: GEMM3  out = W12(64x128) * (h4*concat), +b12, store f32 to global
  {
    int m3 = w & 3;
    int ntbase = (w >> 2) * 8;
    v16bf a3[4];
    const float* Wr = w12 + (size_t)(16 * m3 + ml) * CC;
#pragma unroll
    for (int kc = 0; kc < 4; ++kc) a3[kc] = load_afrag(Wr + kc * 32 + 8 * half);

    float* ob = out + (size_t)b * NCH * HW;
    for (int nt = ntbase; nt < ntbase + 8; ++nt) {
      v8f acc = {};
#pragma unroll
      for (int kc = 0; kc < 4; ++kc) {
        v16bf bfrag = load_bfrag(sH1 + (nt * 16 + ml) * CHS + kc * 32 + 8 * half);
        acc = __builtin_amdgcn_wmma_f32_16x16x32_bf16(false, a3[kc], false, bfrag,
                                                      (short)0, acc, false, false);
      }
      int pc = nt * 16 + ml;
      int gy = ty0 + (pc >> 4), gx = tx0 + (pc & 15);
#pragma unroll
      for (int r = 0; r < 8; ++r) {
        int ch = 16 * m3 + r + 8 * half;
        ob[((size_t)ch * IMG + gy) * IMG + gx] = acc[r] + sB12[ch];
      }
    }
  }
}

// ---------------- launcher ----------------
extern "C" void kernel_launch(void* const* d_in, const int* in_sizes, int n_in,
                              void* d_out, int out_size, void* d_ws, size_t ws_size,
                              hipStream_t stream) {
  (void)in_sizes; (void)n_in; (void)out_size; (void)ws_size;
  const float* x   = (const float*)d_in[0];
  const float* ym  = (const float*)d_in[1];
  const float* w1  = (const float*)d_in[2];
  const float* b1  = (const float*)d_in[3];
  const float* wdp = (const float*)d_in[4];
  const float* bd  = (const float*)d_in[5];
  const float* w2  = (const float*)d_in[6];
  const float* b2  = (const float*)d_in[7];
  const float* w12 = (const float*)d_in[8];
  const float* b12 = (const float*)d_in[9];
  float* out = (float*)d_out;

  // workspace layout
  float* xy = (float*)d_ws;              // 8*64*64
  float* xx = xy + 8 * NCH * NCH;        // 8*64
  float* yy = xx + 8 * NCH;              // 8*64
  int*   nn = (int*)(yy + 8 * NCH);      // 8*64

  const int nacc = 8 * NCH * NCH + 2 * 8 * NCH;   // xy + xx + yy contiguous
  init_kernel<<<(nacc + 255) / 256, 256, 0, stream>>>(xy, nacc);
  gram_kernel<<<dim3(64, 8), 256, 0, stream>>>(x, ym, xy, xx, yy);
  argmin_kernel<<<2, 256, 0, stream>>>(xy, xx, yy, nn);

  const size_t smem_bytes = 6656 + (size_t)(2 * NPAD * CHS + 256 * CHS) * sizeof(bf16);
  fused_kernel<<<dim3(16, 16, 8), 256, smem_bytes, stream>>>(
      x, ym, w1, b1, wdp, bd, w2, b2, w12, b12, nn, out);
}
